// VNDGCNN_44160853738069
// MI455X (gfx1250) — matrix operand order repaired and checked by reading, not compiled, not run on hardware
//
#include <hip/hip_runtime.h>
#include <hip/hip_bf16.h>

typedef __bf16 bf16t;
typedef __attribute__((ext_vector_type(16))) __bf16 v16bf;
typedef __attribute__((ext_vector_type(8)))  __bf16 v8bf;
typedef __attribute__((ext_vector_type(8)))  float   v8f;

#define NB     2
#define NPTS   8192
#define CIN    32
#define FDIM   96      // 32 * 3  (kNN feature dim)
#define KNN    20
#define OCH    64
#define EPSV   1e-6f
#define BNEPS  1e-5f
#define NSLOPE 0.2f

// ---------------------------------------------------------------- helpers ---

__device__ __forceinline__ v8f wmma_bf16(v16bf a, v16bf b, v8f c) {
  // D(f32 16x16) = A(16x32 bf16) * B(32x16 bf16) + C
  return __builtin_amdgcn_wmma_f32_16x16x32_bf16(
      /*neg_a=*/false, a, /*neg_b=*/false, b,
      /*c_mod=*/(short)0, c, /*reuse_a=*/false, /*reuse_b=*/false);
}

// Wave-local LDS producer->consumer fence (cross-lane within one wave32);
// DS ops are in-order per wave, so only the store->load edge needs a wait.
__device__ __forceinline__ void wave_lds_fence() {
  asm volatile("s_wait_dscnt 0" ::: "memory");
}

// ---------------------------------------------------- kernel 0: zero stats --

__global__ void k_zero(float* __restrict__ gstats) {
  if (threadIdx.x < 2 * OCH) gstats[threadIdx.x] = 0.0f;
}

// ------------------------------------- kernel 1: sq-norms + bf16 repacking --
// xbf[b][n][j] (j = c*3+d, 0..95), 192B rows -> 16B-aligned fragment runs.

__global__ void __launch_bounds__(256) k_prep(const float* __restrict__ x,
                                              bf16t* __restrict__ xbf,
                                              float* __restrict__ xx) {
  const int t = blockIdx.x * 256 + threadIdx.x;   // 0 .. NB*NPTS-1
  const int b = t >> 13;
  const int n = t & (NPTS - 1);
  const float* xb = x + (size_t)b * FDIM * NPTS;
  bf16t* orow = xbf + (size_t)t * FDIM;
  float s = 0.0f;
#pragma unroll 8
  for (int j = 0; j < FDIM; ++j) {
    const float v = xb[(size_t)j * NPTS + n];     // coalesced across lanes
    s += v * v;
    orow[j] = (bf16t)v;
  }
  xx[t] = s;
}

// ------------------------------------------- kernel 2: fused kNN via WMMA ---
// Wave: 16 query rows. neg_dist = 2*gram - xx_q - xx_m, Gram via 3 bf16 WMMAs
// (K=96). Unsorted top-20 per lane (set semantics: downstream is permutation
// invariant in k), lane pairs merged at the end.

__global__ void __launch_bounds__(256) k_knn(const bf16t* __restrict__ xbf,
                                             const float* __restrict__ xx,
                                             int* __restrict__ knnOut) {
  __shared__ float tile[8][16][16];
  __shared__ float xxr[8][16];
  __shared__ float mD[8][16][KNN];
  __shared__ int   mI[8][16][KNN];

  const int w    = threadIdx.x >> 5;
  const int lane = threadIdx.x & 31;
  const int half = lane >> 4;
  const int l15  = lane & 15;
  const int b    = blockIdx.x >> 6;                       // 64 blocks / batch
  const int q0   = (blockIdx.x & 63) * 128 + w * 16;
  const int rowp = b * NPTS + q0;

  // A fragments (16 query rows x 96 K) -> 3 k-steps, held in registers.
  v16bf afr[3];
  {
    const bf16t* arow = xbf + (size_t)(rowp + l15) * FDIM;
#pragma unroll
    for (int ks = 0; ks < 3; ++ks) {
      const int kb = ks * 32;
      const v8bf lo = *(const v8bf*)(arow + kb + 8 * half);
      const v8bf hi = *(const v8bf*)(arow + kb + 16 + 8 * half);
#pragma unroll
      for (int e = 0; e < 8; ++e) { afr[ks][e] = lo[e]; afr[ks][8 + e] = hi[e]; }
    }
  }
  if (lane < 16) xxr[w][lane] = xx[rowp + lane];
  wave_lds_fence();

  float bD[KNN]; int bI[KNN];
#pragma unroll
  for (int j = 0; j < KNN; ++j) { bD[j] = -3.0e38f; bI[j] = 0; }
  float minV = -3.0e38f; int minP = 0;

  for (int m0 = 0; m0 < NPTS; m0 += 16) {
    const bf16t* brow = xbf + (size_t)(b * NPTS + m0 + l15) * FDIM;
    v8f acc = {};
#pragma unroll
    for (int ks = 0; ks < 3; ++ks) {
      const int kb = ks * 32;
      v16bf bf;
      const v8bf lo = *(const v8bf*)(brow + kb + 8 * half);
      const v8bf hi = *(const v8bf*)(brow + kb + 16 + 8 * half);
#pragma unroll
      for (int e = 0; e < 8; ++e) { bf[e] = lo[e]; bf[8 + e] = hi[e]; }
      acc = wmma_bf16(afr[ks], bf, acc);
    }
    const float xxc = xx[b * NPTS + m0 + l15];
#pragma unroll
    for (int v = 0; v < 8; ++v) {
      const int r = v + 8 * half;                          // C layout: M=v+8*half
      tile[w][r][l15] = 2.0f * acc[v] - xxr[w][r] - xxc;   //           N=lane&15
    }
    wave_lds_fence();
    // lane scans row l15, columns half*8 .. half*8+7
#pragma unroll
    for (int cc = 0; cc < 8; ++cc) {
      const int col = half * 8 + cc;
      const float vv = tile[w][l15][col];
      if (vv > minV) {
        const int id = m0 + col;
#pragma unroll
        for (int j = 0; j < KNN; ++j) if (j == minP) { bD[j] = vv; bI[j] = id; }
        minV = bD[0]; minP = 0;
#pragma unroll
        for (int j = 1; j < KNN; ++j) if (bD[j] < minV) { minV = bD[j]; minP = j; }
      }
    }
  }

  // merge lane L (cols 0-7) with lane L+16 (cols 8-15)
  if (half == 1) {
#pragma unroll
    for (int j = 0; j < KNN; ++j) { mD[w][l15][j] = bD[j]; mI[w][l15][j] = bI[j]; }
  }
  wave_lds_fence();
  if (half == 0) {
    float cd[2 * KNN]; int ci[2 * KNN];
#pragma unroll
    for (int j = 0; j < KNN; ++j) {
      cd[j] = bD[j];           ci[j] = bI[j];
      cd[KNN + j] = mD[w][l15][j]; ci[KNN + j] = mI[w][l15][j];
    }
    int* outp = knnOut + (size_t)(rowp + l15) * KNN;
    for (int s = 0; s < KNN; ++s) {
      float mx = cd[0]; int mp = 0;
#pragma unroll
      for (int j = 1; j < 2 * KNN; ++j) if (cd[j] > mx) { mx = cd[j]; mp = j; }
      int oid = 0;
#pragma unroll
      for (int j = 0; j < 2 * KNN; ++j) if (j == mp) { oid = ci[j]; cd[j] = -3.4e38f; }
      outp[s] = oid;
    }
  }
}

// ------------------- shared: p = W1 * feat for one wave's 16 (d,k) columns --
// wave owns (pt, tile): point n = pbase+pt, columns col = tile*16 + (lane&15)
// col -> (d = col/20, kk = col%20); cols 60..63 are zero padding.

__device__ __forceinline__ void compute_p_tile(
    const float* __restrict__ x, const float* __restrict__ W,
    const int* __restrict__ knnIn, int b, int n, int tile, int lane,
    v8f cacc[4]) {
  const int half = lane >> 4;
  const int l15  = lane & 15;
  const int col  = tile * 16 + l15;
  const bool valid = (col < 3 * KNN);
  const int d  = valid ? (col / KNN) : 0;
  const int kk = valid ? (col % KNN) : 0;
  const int ni = valid ? knnIn[(size_t)(b * NPTS + n) * KNN + kk] : 0;

  // feat rows c = e<8 ? 8*half + e : 16 + 8*half + (e-8) (+32 for k-step 1); gather ctr/nbr once.
  float ctrv[16], nbrv[16];
  const size_t basec = (size_t)b * CIN * 3 * NPTS;
#pragma unroll
  for (int e = 0; e < 16; ++e) {
    const int c = (e < 8) ? (8 * half + e) : (16 + 8 * half + (e - 8));
    const size_t off = basec + ((size_t)c * 3 + d) * NPTS;
    ctrv[e] = valid ? x[off + n]  : 0.0f;
    nbrv[e] = valid ? x[off + ni] : 0.0f;
  }
  v16bf b0, b1;
#pragma unroll
  for (int e = 0; e < 16; ++e) {
    b0[e] = (bf16t)(nbrv[e] - ctrv[e]);   // channels 0..31 : nbr - ctr
    b1[e] = (bf16t)(ctrv[e]);             // channels 32..63: ctr
  }
#pragma unroll
  for (int mt = 0; mt < 4; ++mt) {        // o-tiles (M = 64)
    v8f c8 = {};
    const float* wrow = W + (size_t)(mt * 16 + l15) * OCH;
#pragma unroll
    for (int ks = 0; ks < 2; ++ks) {      // c contraction (K = 64)
      const int kb = ks * 32;
      v16bf af;
#pragma unroll
      for (int e = 0; e < 8; ++e) {
        af[e]     = (bf16t)wrow[kb + 8 * half + e];
        af[8 + e] = (bf16t)wrow[kb + 16 + 8 * half + e];
      }
      c8 = wmma_bf16(af, (ks == 0) ? b0 : b1, c8);
    }
    cacc[mt] = c8;
  }
}

__device__ __forceinline__ void store_p_lds(float (*pl)[OCH][64], int pt,
                                            int tile, int lane,
                                            const v8f cacc[4]) {
  const int half = lane >> 4, l15 = lane & 15;
#pragma unroll
  for (int mt = 0; mt < 4; ++mt)
#pragma unroll
    for (int v = 0; v < 8; ++v)
      pl[pt][mt * 16 + v + 8 * half][tile * 16 + l15] = cacc[mt][v];
}

// ------------------------- kernel 3: p (recomputable) + VN-BN global stats --

__global__ void __launch_bounds__(256) k_p_stats(
    const float* __restrict__ x, const float* __restrict__ W1,
    const int* __restrict__ knnIn, float* __restrict__ gstats) {
  __shared__ float pl[2][OCH][64];
  __shared__ float sSum[OCH], sSq[OCH];
  const int tid = threadIdx.x;
  const int w = tid >> 5, lane = tid & 31;
  const int b = blockIdx.x / (NPTS / 2);
  const int pbase = (blockIdx.x % (NPTS / 2)) * 2;
  const int pt = w >> 2, tile = w & 3;
  if (tid < OCH) { sSum[tid] = 0.0f; sSq[tid] = 0.0f; }

  v8f cacc[4];
  compute_p_tile(x, W1, knnIn, b, pbase + pt, tile, lane, cacc);
  store_p_lds(pl, pt, tile, lane, cacc);
  __syncthreads();

  for (int t = tid; t < 2 * OCH * KNN; t += 256) {
    const int pt2 = t / (OCH * KNN);
    const int rem = t % (OCH * KNN);
    const int o = rem / KNN, kk = rem % KNN;
    const float p0 = pl[pt2][o][kk];
    const float p1 = pl[pt2][o][KNN + kk];
    const float p2 = pl[pt2][o][2 * KNN + kk];
    const float nrm = sqrtf(p0 * p0 + p1 * p1 + p2 * p2) + EPSV;
    atomicAdd(&sSum[o], nrm);
    atomicAdd(&sSq[o], nrm * nrm);
  }
  __syncthreads();
  if (tid < OCH) {
    atomicAdd(&gstats[tid], sSum[tid]);
    atomicAdd(&gstats[OCH + tid], sSq[tid]);
  }
}

// --------------------------------------------- kernel 4: finalize BN stats --

__global__ void k_stats_final(float* __restrict__ gstats) {
  const int o = threadIdx.x;
  if (o < OCH) {
    const float cnt = (float)(NB * NPTS * KNN);
    const float m = gstats[o] / cnt;
    const float var = gstats[OCH + o] / cnt - m * m;
    gstats[2 * OCH + o] = m;
    gstats[3 * OCH + o] = rsqrtf(var + BNEPS);
  }
}

// -------- kernel 5: recompute p, VN-BN, d = Wd*p', VN-LeakyReLU, mean(k) ---

__global__ void __launch_bounds__(256) k_out(
    const float* __restrict__ x, const float* __restrict__ W1,
    const float* __restrict__ Wd, const float* __restrict__ gamma,
    const float* __restrict__ beta, const int* __restrict__ knnIn,
    const float* __restrict__ gstats, float* __restrict__ out) {
  __shared__ float pl[2][OCH][64];
  __shared__ float dl[2][OCH][64];
  const int tid = threadIdx.x;
  const int w = tid >> 5, lane = tid & 31;
  const int half = lane >> 4, l15 = lane & 15;
  const int b = blockIdx.x / (NPTS / 2);
  const int pbase = (blockIdx.x % (NPTS / 2)) * 2;
  const int pt = w >> 2, tile = w & 3;

  v8f cacc[4];
  compute_p_tile(x, W1, knnIn, b, pbase + pt, tile, lane, cacc);
  store_p_lds(pl, pt, tile, lane, cacc);
  __syncthreads();

  // VN-BatchNorm, in place
  for (int t = tid; t < 2 * OCH * KNN; t += 256) {
    const int pt2 = t / (OCH * KNN);
    const int rem = t % (OCH * KNN);
    const int o = rem / KNN, kk = rem % KNN;
    const float p0 = pl[pt2][o][kk];
    const float p1 = pl[pt2][o][KNN + kk];
    const float p2 = pl[pt2][o][2 * KNN + kk];
    const float nrm = sqrtf(p0 * p0 + p1 * p1 + p2 * p2) + EPSV;
    const float f = (gamma[o] * (nrm - gstats[2 * OCH + o]) * gstats[3 * OCH + o]
                     + beta[o]) / nrm;
    pl[pt2][o][kk]           = p0 * f;
    pl[pt2][o][KNN + kk]     = p1 * f;
    pl[pt2][o][2 * KNN + kk] = p2 * f;
  }
  __syncthreads();

  // d = Wd * p'  (B fragments re-read from LDS)
  {
    v16bf b0, b1;
#pragma unroll
    for (int e = 0; e < 16; ++e) {
      b0[e] = (bf16t)pl[pt][16 * half + e][tile * 16 + l15];
      b1[e] = (bf16t)pl[pt][32 + 16 * half + e][tile * 16 + l15];
    }
    v8f dacc[4];
#pragma unroll
    for (int mt = 0; mt < 4; ++mt) {
      v8f c8 = {};
      const float* wrow = Wd + (size_t)(mt * 16 + l15) * OCH;
#pragma unroll
      for (int ks = 0; ks < 2; ++ks) {
        const int kb = ks * 32;
        v16bf af;
#pragma unroll
        for (int e = 0; e < 8; ++e) {
          af[e]     = (bf16t)wrow[kb + 16 * half + e];
          af[8 + e] = (bf16t)wrow[kb + 16 * half + 8 + e];
        }
        c8 = wmma_bf16(af, (ks == 0) ? b0 : b1, c8);
      }
      dacc[mt] = c8;
    }
    store_p_lds(dl, pt, tile, lane, dacc);
  }
  __syncthreads();

  // VN-LeakyReLU projection + mean over k
  for (int t = tid; t < 2 * OCH * 3; t += 256) {
    const int pt2 = t / (OCH * 3);
    const int rem = t % (OCH * 3);
    const int o = rem / 3, d = rem % 3;
    float acc = 0.0f;
    for (int kk = 0; kk < KNN; ++kk) {
      float dot = 0.0f, dn = 0.0f;
#pragma unroll
      for (int d2 = 0; d2 < 3; ++d2) {
        const float a = pl[pt2][o][d2 * KNN + kk];
        const float g = dl[pt2][o][d2 * KNN + kk];
        dot += a * g; dn += g * g;
      }
      const float pv = pl[pt2][o][d * KNN + kk];
      const float dv = dl[pt2][o][d * KNN + kk];
      const float proj = pv - dot / (dn + EPSV) * dv;
      acc += NSLOPE * pv + (1.0f - NSLOPE) * ((dot >= 0.0f) ? pv : proj);
    }
    out[(((size_t)b * OCH + o) * 3 + d) * NPTS + (pbase + pt2)] =
        acc * (1.0f / KNN);
  }
}

// ---------------------------------------------------------------- launcher --

extern "C" void kernel_launch(void* const* d_in, const int* in_sizes, int n_in,
                              void* d_out, int out_size, void* d_ws,
                              size_t ws_size, hipStream_t stream) {
  (void)in_sizes; (void)n_in; (void)out_size; (void)ws_size;
  const float* x     = (const float*)d_in[0];
  const float* W1    = (const float*)d_in[1];
  const float* Wd    = (const float*)d_in[2];
  const float* gamma = (const float*)d_in[3];
  const float* beta  = (const float*)d_in[4];
  float* out = (float*)d_out;

  // workspace layout (~4.52 MB total)
  char* ws = (char*)d_ws;
  bf16t* xbf = (bf16t*)ws;                               // 3,145,728 B
  float* xx  = (float*)(ws + 3145728);                   //    65,536 B
  int*   knn = (int*)(ws + 3145728 + 65536);             // 1,310,720 B
  float* gst = (float*)(ws + 3145728 + 65536 + 1310720); //     1,024 B

  k_zero       <<<1, 256, 0, stream>>>(gst);
  k_prep       <<<(NB * NPTS) / 256, 256, 0, stream>>>(x, xbf, xx);
  k_knn        <<<NB * (NPTS / 128), 256, 0, stream>>>(xbf, xx, knn);
  k_p_stats    <<<NB * (NPTS / 2), 256, 0, stream>>>(x, W1, knn, gst);
  k_stats_final<<<1, 64, 0, stream>>>(gst);
  k_out        <<<NB * (NPTS / 2), 256, 0, stream>>>(x, W1, Wd, gamma, beta,
                                                     knn, gst, out);
}
